// MolecularDiffusionModel_73993696575518
// MI455X (gfx1250) — matrix operand order repaired and checked by reading, not compiled, not run on hardware
//
#include <hip/hip_runtime.h>
#include <hip/hip_bf16.h>

#define HD      256
#define NLAYERS 8

// Tiling: 64 rows (edges/nodes) per block, 256 threads = 8 waves,
// each wave owns 2 N-tiles x 4 M-tiles.
#define MROWS   64
#define LDF     776   // feature row stride (768 + 8 pad) in bf16 elems
#define LDA     264   // activation row stride (256 + 8 pad) in bf16 elems
#define SRS     260   // fp32 LN-residual row stride in floats

#define ACTA_OFF  0                         // bytes into dynamic LDS
#define ACTB_OFF  (MROWS * LDA * 2)         // 33792
#define SRES_OFF  (MROWS * LDA * 2)         // node kernel: fp32 buf after actA
#define SMEM_EDGE (MROWS * LDF * 2)                         // 99328 B
#define SMEM_NODE (MROWS * LDA * 2 + MROWS * SRS * 4)       // 100352 B

typedef __attribute__((ext_vector_type(16))) __bf16 v16bf;
typedef __attribute__((ext_vector_type(8)))  __bf16 v8bf;
typedef __attribute__((ext_vector_type(8)))  float  v8f;

__device__ __forceinline__ float silu_f(float x) {
    return x / (1.0f + __expf(-x));
}

__device__ __forceinline__ v8f v8f_zero() {
    v8f z = {0.f, 0.f, 0.f, 0.f, 0.f, 0.f, 0.f, 0.f};
    return z;
}

// ---------------------------------------------------------------------------
// Weight fragment pre-swizzle: fp32 [L][src_rows][256] row-major ->
// bf16 fragments [L][nt=16][kb=KB][lane=32][j=16]; one contiguous 32B load
// per lane per B fragment in the GEMM.
// B k order used here (contiguous, same as A): lane L holds column (L&15),
//   element j <-> k = kb*32 + (L&16 ? 16 : 0) + j.
// skip512: e_w1 dist row (k=512) excluded from WMMA K; handled as rank-1.
// ---------------------------------------------------------------------------
__global__ void wfrag_kernel(const float* __restrict__ src, __bf16* __restrict__ dst,
                             int KB, int src_rows, int skip512) {
    int idx = blockIdx.x * blockDim.x + threadIdx.x;
    int total = NLAYERS * 16 * KB * 32;
    if (idx >= total) return;
    int lane = idx & 31;
    int t    = idx >> 5;
    int kb   = t % KB;  t /= KB;
    int nt   = t & 15;  t >>= 4;
    int layer = t;
    int n     = nt * 16 + (lane & 15);
    int kbase = kb * 32 + ((lane & 16) ? 16 : 0);
    __bf16* out = dst + (size_t)idx * 16;
    const float* s = src + (size_t)layer * src_rows * HD;
#pragma unroll
    for (int j = 0; j < 16; ++j) {
        int k = kbase + j;
        if (skip512 && k >= 512) k += 1;
        out[j] = (__bf16)s[(size_t)k * HD + n];
    }
}

// ---------------------------------------------------------------------------
// Multi-tile WMMA core: 64-row tile in LDS (row stride ldA, bf16) times the
// 256-column weight, K = KB*32.  Per k-block: 4 A fragments (ds_load_b128 x2
// each), 2 B fragments per wave (one 32B global load per lane), 8 WMMAs.
// A k order used here (contiguous, same as B): lane L holds row (L&15),
//   element j <-> k = j + (L&16 ? 16 : 0).
// ---------------------------------------------------------------------------
__device__ __forceinline__ void wmma_multitile(const __bf16* sA, int ldA,
                                               const __bf16* wf, int KB,
                                               int layer, int lane, int wave,
                                               v8f acc[2][4]) {
    const int m    = lane & 15;
    const int hsel = (lane & 16) ? 16 : 0;
    for (int kb = 0; kb < KB; ++kb) {
        v16bf a[4];
#pragma unroll
        for (int mt = 0; mt < 4; ++mt) {
            const __bf16* ap = sA + (size_t)(mt * 16 + m) * ldA + kb * 32 + hsel;
            v8bf lo = *(const v8bf*)(ap);        // k = base+0..7
            v8bf hi = *(const v8bf*)(ap + 8);    // k = base+8..15
            a[mt] = __builtin_shufflevector(lo, hi,
                        0,1,2,3,4,5,6,7,8,9,10,11,12,13,14,15);
        }
#pragma unroll
        for (int t = 0; t < 2; ++t) {
            int nt = wave * 2 + t;
            v16bf b = *(const v16bf*)(wf +
                ((((size_t)layer * 16 + nt) * KB + kb) * 32 + lane) * 16);
#pragma unroll
            for (int mt = 0; mt < 4; ++mt) {
                acc[t][mt] = __builtin_amdgcn_wmma_f32_16x16x32_bf16(
                    false, a[mt], false, b, (short)0, acc[t][mt], false, false);
            }
        }
    }
}

// ---------------------------------------------------------------------------
// Time embedding: sinusoidal -> 2-layer MLP.  B=64 graphs, trivial cost.
// ---------------------------------------------------------------------------
__global__ void time_embed_kernel(const int* __restrict__ ts,
                                  const float* __restrict__ tw1, const float* __restrict__ tb1,
                                  const float* __restrict__ tw2, const float* __restrict__ tb2,
                                  float* __restrict__ temb) {
    __shared__ float s0[HD];
    __shared__ float s1[HD];
    int b = blockIdx.x;
    int n = threadIdx.x;
    float t = (float)ts[b];
    if (n < HD / 2) {
        float fr = __expf(-9.210340371976184f * (float)n / (float)(HD / 2)); // ln(1e4)
        float a  = t * fr;
        s0[n]          = __sinf(a);
        s0[n + HD / 2] = __cosf(a);
    }
    __syncthreads();
    float acc = tb1[n];
    for (int k = 0; k < HD; ++k) acc = fmaf(s0[k], tw1[k * HD + n], acc);
    s1[n] = silu_f(acc);
    __syncthreads();
    float acc2 = tb2[n];
    for (int k = 0; k < HD; ++k) acc2 = fmaf(s1[k], tw2[k * HD + n], acc2);
    temb[b * HD + n] = acc2;
}

__global__ void hinit_kernel(const int* __restrict__ at, const float* __restrict__ emb,
                             float* __restrict__ h, int N) {
    int idx = blockIdx.x * blockDim.x + threadIdx.x;
    if (idx < N * HD) {
        int node = idx >> 8;
        int f    = idx & 255;
        h[idx] = emb[(size_t)at[node] * HD + f];
    }
}

// ---------------------------------------------------------------------------
// Edge kernel: one block = 64 edges, 256 threads.  Full edge-MLP chain in
// WMMA with fused atomic segment-sums (msg, pos delta).  The big LDS staging
// buffer is re-used as the activation ping-pong (extra barriers make the
// aliasing safe); m[E,256] is never materialized.
// ---------------------------------------------------------------------------
__global__ __launch_bounds__(256) void edge_kernel(
    const float* __restrict__ h, const float* __restrict__ pos_cur,
    float* __restrict__ pos_next, float* __restrict__ msg,
    const float* __restrict__ temb, const int* __restrict__ ei,
    const int* __restrict__ batch,
    const __bf16* __restrict__ ew1f, const float* __restrict__ ew1full,
    const float* __restrict__ eb1,
    const __bf16* __restrict__ ew2f, const float* __restrict__ eb2,
    const __bf16* __restrict__ cw1f, const float* __restrict__ cb1,
    const float* __restrict__ cw2,
    int layer, int E) {
    extern __shared__ char smem[];
    __bf16* sfeat = (__bf16*)smem;                 // [64][LDF]
    __bf16* actA  = (__bf16*)(smem + ACTA_OFF);    // [64][LDA] (aliases sfeat)
    __bf16* actB  = (__bf16*)(smem + ACTB_OFF);    // [64][LDA]

    __shared__ float sdiff[MROWS * 3];
    __shared__ float sdist[MROWS];
    __shared__ int   srow[MROWS], scol[MROWS], sbr[MROWS];

    const int tid  = threadIdx.x;
    const int lane = tid & 31;
    const int wave = tid >> 5;
    const int e0   = blockIdx.x * MROWS;

    if (tid < MROWS) {
        int e = e0 + tid;
        int r = ei[e];
        int c = ei[E + e];
        srow[tid] = r; scol[tid] = c; sbr[tid] = batch[r];
        float dx = pos_cur[r * 3 + 0] - pos_cur[c * 3 + 0];
        float dy = pos_cur[r * 3 + 1] - pos_cur[c * 3 + 1];
        float dz = pos_cur[r * 3 + 2] - pos_cur[c * 3 + 2];
        sdiff[tid * 3 + 0] = dx; sdiff[tid * 3 + 1] = dy; sdiff[tid * 3 + 2] = dz;
        sdist[tid] = sqrtf(dx * dx + dy * dy + dz * dz + 1e-8f);
    }
    __syncthreads();

    for (int i = tid; i < MROWS * 768; i += 256) {
        int e = i / 768;
        int f = i - e * 768;
        float v;
        if (f < HD)          v = h[(size_t)srow[e] * HD + f];
        else if (f < 2 * HD) v = h[(size_t)scol[e] * HD + (f - HD)];
        else                 v = temb[(size_t)sbr[e] * HD + (f - 2 * HD)];
        sfeat[(size_t)e * LDF + f] = (__bf16)v;
    }
    __syncthreads();

    const int m8 = (lane & 16) ? 8 : 0;

    // GEMM1: edge_feat[64x768] @ e_w1 (+ dist rank-1 + bias), SiLU -> actA
    {
        v8f acc[2][4];
#pragma unroll
        for (int t = 0; t < 2; ++t)
#pragma unroll
            for (int mt = 0; mt < 4; ++mt) acc[t][mt] = v8f_zero();
        wmma_multitile(sfeat, LDF, ew1f, 24, layer, lane, wave, acc);
        __syncthreads();   // all waves done reading sfeat; actA may alias it
#pragma unroll
        for (int t = 0; t < 2; ++t) {
            int n    = (wave * 2 + t) * 16 + (lane & 15);
            float wd = ew1full[((size_t)layer * 769 + 512) * HD + n]; // dist row
            float bb = eb1[layer * HD + n];
#pragma unroll
            for (int mt = 0; mt < 4; ++mt)
#pragma unroll
                for (int v = 0; v < 8; ++v) {
                    int row = mt * 16 + v + m8;
                    float x = acc[t][mt][v] + sdist[row] * wd + bb;
                    actA[(size_t)row * LDA + n] = (__bf16)silu_f(x);
                }
        }
    }
    __syncthreads();

    // GEMM2: m = silu(actA @ e_w2 + b) -> actB, fused msg segment-sum
    {
        v8f acc[2][4];
#pragma unroll
        for (int t = 0; t < 2; ++t)
#pragma unroll
            for (int mt = 0; mt < 4; ++mt) acc[t][mt] = v8f_zero();
        wmma_multitile(actA, LDA, ew2f, 8, layer, lane, wave, acc);
#pragma unroll
        for (int t = 0; t < 2; ++t) {
            int n    = (wave * 2 + t) * 16 + (lane & 15);
            float bb = eb2[layer * HD + n];
#pragma unroll
            for (int mt = 0; mt < 4; ++mt)
#pragma unroll
                for (int v = 0; v < 8; ++v) {
                    int row = mt * 16 + v + m8;
                    float x = silu_f(acc[t][mt][v] + bb);
                    unsafeAtomicAdd(&msg[(size_t)srow[row] * HD + n], x);
                    actB[(size_t)row * LDA + n] = (__bf16)x;
                }
        }
    }
    __syncthreads();

    // GEMM3: p = silu(m @ c_w1 + b) -> actA (actA's last reader was GEMM2)
    {
        v8f acc[2][4];
#pragma unroll
        for (int t = 0; t < 2; ++t)
#pragma unroll
            for (int mt = 0; mt < 4; ++mt) acc[t][mt] = v8f_zero();
        wmma_multitile(actB, LDA, cw1f, 8, layer, lane, wave, acc);
#pragma unroll
        for (int t = 0; t < 2; ++t) {
            int n    = (wave * 2 + t) * 16 + (lane & 15);
            float bb = cb1[layer * HD + n];
#pragma unroll
            for (int mt = 0; mt < 4; ++mt)
#pragma unroll
                for (int v = 0; v < 8; ++v) {
                    int row = mt * 16 + v + m8;
                    actA[(size_t)row * LDA + n] = (__bf16)silu_f(acc[t][mt][v] + bb);
                }
        }
    }
    __syncthreads();

    // coord scalar = p . c_w2 ; scatter coord*unit into pos_next
    if (tid < MROWS) {
        const float* w = cw2 + (size_t)layer * HD;
        float acc = 0.f;
        for (int k = 0; k < HD; ++k)
            acc = fmaf((float)actA[(size_t)tid * LDA + k], w[k], acc);
        float inv = acc / (sdist[tid] + 1e-8f);
        int r = srow[tid];
        unsafeAtomicAdd(&pos_next[r * 3 + 0], sdiff[tid * 3 + 0] * inv);
        unsafeAtomicAdd(&pos_next[r * 3 + 1], sdiff[tid * 3 + 1] * inv);
        unsafeAtomicAdd(&pos_next[r * 3 + 2], sdiff[tid * 3 + 2] * inv);
    }
}

// ---------------------------------------------------------------------------
// Node kernel: one block = 64 nodes.  concat[h,msg,t] MLP + residual + LN.
// In-place h update is safe: each node belongs to exactly one block.
// ---------------------------------------------------------------------------
__global__ __launch_bounds__(256) void node_kernel(
    float* __restrict__ h, const float* __restrict__ msg,
    const float* __restrict__ temb, const int* __restrict__ batch,
    const __bf16* __restrict__ nw1f, const float* __restrict__ nb1,
    const __bf16* __restrict__ nw2f, const float* __restrict__ nb2,
    const float* __restrict__ lng, const float* __restrict__ lnb,
    int layer, int N) {
    extern __shared__ char smem[];
    __bf16* sfeat = (__bf16*)smem;                 // [64][LDF]
    __bf16* actA  = (__bf16*)(smem + ACTA_OFF);    // [64][LDA] (aliases sfeat)
    float*  sres  = (float*)(smem + SRES_OFF);     // [64][SRS] fp32

    __shared__ float smean[MROWS], srstd[MROWS];

    const int tid  = threadIdx.x;
    const int lane = tid & 31;
    const int wave = tid >> 5;
    const int n0   = blockIdx.x * MROWS;

    for (int i = tid; i < MROWS * 768; i += 256) {
        int e    = i / 768;
        int f    = i - e * 768;
        int node = n0 + e;
        float v = 0.f;
        if (node < N) {
            if (f < HD)          v = h[(size_t)node * HD + f];
            else if (f < 2 * HD) v = msg[(size_t)node * HD + (f - HD)];
            else                 v = temb[(size_t)batch[node] * HD + (f - 2 * HD)];
        }
        sfeat[(size_t)e * LDF + f] = (__bf16)v;
    }
    __syncthreads();

    const int m8 = (lane & 16) ? 8 : 0;

    {   // GEMM1: [64x768] @ n_w1 + b, SiLU -> actA
        v8f acc[2][4];
#pragma unroll
        for (int t = 0; t < 2; ++t)
#pragma unroll
            for (int mt = 0; mt < 4; ++mt) acc[t][mt] = v8f_zero();
        wmma_multitile(sfeat, LDF, nw1f, 24, layer, lane, wave, acc);
        __syncthreads();   // sfeat dead; actA aliases it
#pragma unroll
        for (int t = 0; t < 2; ++t) {
            int n    = (wave * 2 + t) * 16 + (lane & 15);
            float bb = nb1[layer * HD + n];
#pragma unroll
            for (int mt = 0; mt < 4; ++mt)
#pragma unroll
                for (int v = 0; v < 8; ++v) {
                    int row = mt * 16 + v + m8;
                    actA[(size_t)row * LDA + n] = (__bf16)silu_f(acc[t][mt][v] + bb);
                }
        }
    }
    __syncthreads();

    {   // GEMM2: @ n_w2 + b, + residual h -> sres (fp32)
        v8f acc[2][4];
#pragma unroll
        for (int t = 0; t < 2; ++t)
#pragma unroll
            for (int mt = 0; mt < 4; ++mt) acc[t][mt] = v8f_zero();
        wmma_multitile(actA, LDA, nw2f, 8, layer, lane, wave, acc);
#pragma unroll
        for (int t = 0; t < 2; ++t) {
            int n    = (wave * 2 + t) * 16 + (lane & 15);
            float bb = nb2[layer * HD + n];
#pragma unroll
            for (int mt = 0; mt < 4; ++mt)
#pragma unroll
                for (int v = 0; v < 8; ++v) {
                    int row  = mt * 16 + v + m8;
                    int node = n0 + row;
                    float hv = (node < N) ? h[(size_t)node * HD + n] : 0.f;
                    sres[(size_t)row * SRS + n] = acc[t][mt][v] + bb + hv;
                }
        }
    }
    __syncthreads();

    if (tid < MROWS && (n0 + tid) < N) {
        float mn = 0.f;
        for (int k = 0; k < HD; ++k) mn += sres[(size_t)tid * SRS + k];
        mn *= (1.0f / HD);
        float vr = 0.f;
        for (int k = 0; k < HD; ++k) {
            float d = sres[(size_t)tid * SRS + k] - mn;
            vr = fmaf(d, d, vr);
        }
        vr *= (1.0f / HD);
        smean[tid] = mn;
        srstd[tid] = rsqrtf(vr + 1e-5f);
    }
    __syncthreads();

    for (int i = tid; i < MROWS * HD; i += 256) {
        int e = i >> 8;
        int f = i & 255;
        int node = n0 + e;
        if (node < N) {
            h[(size_t)node * HD + f] =
                (sres[(size_t)e * SRS + f] - smean[e]) * srstd[e]
                * lng[layer * HD + f] + lnb[layer * HD + f];
        }
    }
}

// ---------------------------------------------------------------------------
// Output heads: coords [N,3] then atoms [N,10], concatenated flat.
// ---------------------------------------------------------------------------
__global__ void final_kernel(const float* __restrict__ h,
                             const float* __restrict__ cw, const float* __restrict__ cb,
                             const float* __restrict__ aw, const float* __restrict__ ab,
                             float* __restrict__ out, int N) {
    int idx = blockIdx.x * blockDim.x + threadIdx.x;
    int n3  = N * 3;
    if (idx >= N * 13) return;
    if (idx < n3) {
        int node = idx / 3, c = idx % 3;
        float acc = cb[c];
        const float* hr = h + (size_t)node * HD;
        for (int k = 0; k < HD; ++k) acc = fmaf(hr[k], cw[k * 3 + c], acc);
        out[idx] = acc;
    } else {
        int o = idx - n3;
        int node = o / 10, c = o % 10;
        float acc = ab[c];
        const float* hr = h + (size_t)node * HD;
        for (int k = 0; k < HD; ++k) acc = fmaf(hr[k], aw[k * 10 + c], acc);
        out[idx] = acc;
    }
}

// ---------------------------------------------------------------------------
extern "C" void kernel_launch(void* const* d_in, const int* in_sizes, int n_in,
                              void* d_out, int out_size, void* d_ws, size_t ws_size,
                              hipStream_t stream) {
    const int*   at    = (const int*)d_in[0];
    const float* pos   = (const float*)d_in[1];
    const int*   ei    = (const int*)d_in[2];
    const int*   ts    = (const int*)d_in[3];
    const int*   batch = (const int*)d_in[4];
    const float* emb   = (const float*)d_in[5];
    const float* tw1   = (const float*)d_in[6];
    const float* tb1   = (const float*)d_in[7];
    const float* tw2   = (const float*)d_in[8];
    const float* tb2   = (const float*)d_in[9];
    const float* ew1   = (const float*)d_in[10];
    const float* eb1   = (const float*)d_in[11];
    const float* ew2   = (const float*)d_in[12];
    const float* eb2   = (const float*)d_in[13];
    const float* nw1   = (const float*)d_in[14];
    const float* nb1   = (const float*)d_in[15];
    const float* nw2   = (const float*)d_in[16];
    const float* nb2   = (const float*)d_in[17];
    const float* cw1   = (const float*)d_in[18];
    const float* cb1   = (const float*)d_in[19];
    const float* cw2   = (const float*)d_in[20];
    const float* lng   = (const float*)d_in[21];
    const float* lnb   = (const float*)d_in[22];
    const float* cow   = (const float*)d_in[23];
    const float* cob   = (const float*)d_in[24];
    const float* aw    = (const float*)d_in[25];
    const float* ab    = (const float*)d_in[26];

    const int N = in_sizes[0];
    const int E = in_sizes[2] / 2;
    const int B = in_sizes[3];
    (void)n_in; (void)out_size; (void)ws_size;

    size_t off = 0;
    auto carve = [&](size_t bytes) -> void* {
        void* p = (char*)d_ws + off;
        off += (bytes + 255) & ~(size_t)255;
        return p;
    };
    float*  h    = (float*)carve((size_t)N * HD * 4);
    float*  msg  = (float*)carve((size_t)N * HD * 4);
    float*  posA = (float*)carve((size_t)N * 3 * 4);
    float*  posB = (float*)carve((size_t)N * 3 * 4);
    float*  temb = (float*)carve((size_t)B * HD * 4);
    __bf16* ew1f = (__bf16*)carve((size_t)NLAYERS * 16 * 24 * 32 * 16 * 2);
    __bf16* ew2f = (__bf16*)carve((size_t)NLAYERS * 16 * 8  * 32 * 16 * 2);
    __bf16* cw1f = (__bf16*)carve((size_t)NLAYERS * 16 * 8  * 32 * 16 * 2);
    __bf16* nw1f = (__bf16*)carve((size_t)NLAYERS * 16 * 24 * 32 * 16 * 2);
    __bf16* nw2f = (__bf16*)carve((size_t)NLAYERS * 16 * 8  * 32 * 16 * 2);

    // Prep: time MLP, h init, weight -> bf16 fragment swizzle
    time_embed_kernel<<<B, HD, 0, stream>>>(ts, tw1, tb1, tw2, tb2, temb);
    hinit_kernel<<<(N * HD + 255) / 256, 256, 0, stream>>>(at, emb, h, N);
    const int g24 = NLAYERS * 16 * 24 * 32;
    const int g8  = NLAYERS * 16 * 8  * 32;
    wfrag_kernel<<<(g24 + 255) / 256, 256, 0, stream>>>(ew1, ew1f, 24, 769, 1);
    wfrag_kernel<<<(g8  + 255) / 256, 256, 0, stream>>>(ew2, ew2f, 8, 256, 0);
    wfrag_kernel<<<(g8  + 255) / 256, 256, 0, stream>>>(cw1, cw1f, 8, 256, 0);
    wfrag_kernel<<<(g24 + 255) / 256, 256, 0, stream>>>(nw1, nw1f, 24, 768, 0);
    wfrag_kernel<<<(g8  + 255) / 256, 256, 0, stream>>>(nw2, nw2f, 8, 256, 0);

    hipMemcpyAsync(posA, pos, (size_t)N * 3 * 4, hipMemcpyDeviceToDevice, stream);

    const int eblocks = E / MROWS;
    const int nblocks = (N + MROWS - 1) / MROWS;
    for (int layer = 0; layer < NLAYERS; ++layer) {
        float* pc = (layer & 1) ? posB : posA;
        float* pn = (layer & 1) ? posA : posB;
        hipMemcpyAsync(pn, pc, (size_t)N * 3 * 4, hipMemcpyDeviceToDevice, stream);
        hipMemsetAsync(msg, 0, (size_t)N * HD * 4, stream);
        edge_kernel<<<eblocks, 256, SMEM_EDGE, stream>>>(h, pc, pn, msg, temb, ei, batch,
                                                         ew1f, ew1, eb1, ew2f, eb2,
                                                         cw1f, cb1, cw2, layer, E);
        node_kernel<<<nblocks, 256, SMEM_NODE, stream>>>(h, msg, temb, batch,
                                                         nw1f, nb1, nw2f, nb2,
                                                         lng, lnb, layer, N);
    }

    final_kernel<<<(N * 13 + 255) / 256, 256, 0, stream>>>(h, cow, cob, aw, ab,
                                                           (float*)d_out, N);
}
